// AttGRU_42820823941244
// MI455X (gfx1250) — matrix operand order repaired and checked by reading, not compiled, not run on hardware
//
#include <hip/hip_runtime.h>

// Problem dims (fixed by the reference)
#define Bn 16
#define Ss 64
#define Nn 2048
#define Ee 256

typedef __attribute__((ext_vector_type(16))) __bf16 v16bf;
typedef __attribute__((ext_vector_type(8)))  float  v8f;

// ---------------------------------------------------------------------------
// Fragment loaders per CDNA5 ISA VGPR layouts (CDNA5 ISA §7.12.2)
// A (16x32 bf16): lanes 0-15 row m=lane, k = {0..7,16..23}; lanes 16-31 same
// rows, k = {8..15,24..31}.  => two contiguous 8-elem chunks per lane.
// B (32x16 bf16): lanes 0-15 col n=lane, k = {0..7,16..23}; lanes 16-31, k = {8..15,24..31}.
// C/D (16x16 f32): vgpr r, lane l -> element (m = r + 8*(l>>4), n = l&15).
// ---------------------------------------------------------------------------
__device__ __forceinline__ v16bf load_a_bf16(const __bf16* rowk, int half) {
    const __bf16* p = rowk + 8 * half;
    v16bf f;
#pragma unroll
    for (int i = 0; i < 8; ++i) f[i] = p[i];
#pragma unroll
    for (int i = 0; i < 8; ++i) f[8 + i] = p[16 + i];
    return f;
}

__device__ __forceinline__ v16bf load_b_bf16(const __bf16* colk, int half) {
    const __bf16* p = colk + 8 * half;
    v16bf f;
#pragma unroll
    for (int i = 0; i < 8; ++i) f[i] = p[i];
#pragma unroll
    for (int i = 0; i < 8; ++i) f[8 + i] = p[16 + i];
    return f;
}

__device__ __forceinline__ v16bf load_b_f32cvt(const float* colk, int half) {
    const float* p = colk + 8 * half;
    v16bf f;
#pragma unroll
    for (int i = 0; i < 8; ++i) f[i] = (__bf16)p[i];
#pragma unroll
    for (int i = 0; i < 8; ++i) f[8 + i] = (__bf16)p[16 + i];
    return f;
}

#define WMMA_BF16(a, b, c) \
    __builtin_amdgcn_wmma_f32_16x16x32_bf16(false, (a), false, (b), (short)0, (c), false, false)

// ---------------------------------------------------------------------------
// Kernel 0: convert GRU weights fp32 -> bf16, zero h0 (fp32 + bf16)
// ---------------------------------------------------------------------------
__global__ void attgru_prep_kernel(const float* Whr, const float* Whz, const float* Whn,
                                   __bf16* Wr, __bf16* Wz, __bf16* Wn,
                                   float* hfp0, __bf16* hbf0) {
    size_t stride = (size_t)gridDim.x * blockDim.x;
    size_t idx = (size_t)blockIdx.x * blockDim.x + threadIdx.x;
    size_t total = (size_t)Nn * Nn;
    for (size_t i = idx; i < total; i += stride) {
        Wr[i] = (__bf16)Whr[i];
        Wz[i] = (__bf16)Whz[i];
        Wn[i] = (__bf16)Whn[i];
    }
    for (size_t i = idx; i < (size_t)Bn * Nn; i += stride) {
        hfp0[i] = 0.0f;
        hbf0[i] = (__bf16)0.0f;
    }
}

// ---------------------------------------------------------------------------
// Kernel 1: Q/K projection.  Q[b,n,e] = sum_s x[b,s,n]*Wq[e,s] + bq[e] (bf16 out)
// grid: (Bn*Nn/16, Ee/16, 2[q|k]); block: 32 (one wave)
// ---------------------------------------------------------------------------
__global__ __launch_bounds__(32) void attgru_proj_kernel(
        const float* x, const float* Wq, const float* bq,
        const float* Wk, const float* bk, __bf16* Qbf, __bf16* Kbf) {
    int lane = threadIdx.x, half = lane >> 4, l16 = lane & 15;
    int b  = blockIdx.x / (Nn / 16);
    int n0 = (blockIdx.x % (Nn / 16)) * 16;
    int e0 = blockIdx.y * 16;
    const float* W    = blockIdx.z ? Wk : Wq;
    const float* bias = blockIdx.z ? bk : bq;
    __bf16*      out  = blockIdx.z ? Kbf : Qbf;

    v8f acc = {};
    const float* Bcol = W + (size_t)(e0 + l16) * Ss;      // weight row = B column
    for (int kb = 0; kb < Ss; kb += 32) {
        // A[m][k=s] = x[b][kb+k][n0+m]  (stride Nn in k)
        const float* ab = x + ((size_t)b * Ss + kb) * Nn + (n0 + l16);
        v16bf a;
#pragma unroll
        for (int i = 0; i < 8; ++i) a[i]     = (__bf16)ab[(size_t)(8 * half + i) * Nn];
#pragma unroll
        for (int i = 0; i < 8; ++i) a[8 + i] = (__bf16)ab[(size_t)(16 + 8 * half + i) * Nn];
        v16bf bf = load_b_f32cvt(Bcol + kb, half);
        acc = WMMA_BF16(a, bf, acc);
    }
#pragma unroll
    for (int r = 0; r < 8; ++r) {
        int mm = r + 8 * half;                  // node row in tile
        int ee = e0 + l16;                      // embed col
        float v = acc[r] + bias[ee];
        out[((size_t)b * Nn + n0 + mm) * Ee + ee] = (__bf16)v;
    }
}

// ---------------------------------------------------------------------------
// Kernel 2: scores[b,n,m] = Q[b,n,:] . K[b,m,:], masked by adj
// grid: (Nn/16, Nn/16, Bn); block: 32
// ---------------------------------------------------------------------------
__global__ __launch_bounds__(32) void attgru_scores_kernel(
        const __bf16* Qbf, const __bf16* Kbf, const int* adj, float* scores) {
    int lane = threadIdx.x, half = lane >> 4, l16 = lane & 15;
    int n0 = blockIdx.x * 16;
    int m0 = blockIdx.y * 16;
    int b  = blockIdx.z;
    const __bf16* Arow = Qbf + ((size_t)b * Nn + n0 + l16) * Ee;
    const __bf16* Bcol = Kbf + ((size_t)b * Nn + m0 + l16) * Ee;   // K row = B column
    v8f acc = {};
    for (int kb = 0; kb < Ee; kb += 32) {
        v16bf a  = load_a_bf16(Arow + kb, half);
        v16bf bf = load_b_bf16(Bcol + kb, half);
        acc = WMMA_BF16(a, bf, acc);
    }
#pragma unroll
    for (int r = 0; r < 8; ++r) {
        int nn = n0 + r + 8 * half;
        int mm = m0 + l16;
        float v = (adj[(size_t)nn * Nn + mm] > 0) ? acc[r] : -9.0e15f;
        scores[((size_t)b * Nn + nn) * Nn + mm] = v;
    }
}

// ---------------------------------------------------------------------------
// Kernel 3: column-wise (axis=1 / over n) max + sumexp per (b,m)
// grid: Bn*(Nn/256); block: 256 (coalesced along m)
// ---------------------------------------------------------------------------
__global__ void attgru_colreduce_kernel(const float* scores, float* colmax, float* colsum) {
    int b = blockIdx.x / (Nn / 256);
    int m = (blockIdx.x % (Nn / 256)) * 256 + threadIdx.x;
    const float* col = scores + (size_t)b * Nn * Nn + m;
    float mx = -3.4e38f;
    for (int n = 0; n < Nn; ++n) mx = fmaxf(mx, col[(size_t)n * Nn]);
    float sm = 0.0f;
    for (int n = 0; n < Nn; ++n) sm += __expf(col[(size_t)n * Nn] - mx);
    colmax[(size_t)b * Nn + m] = mx;
    colsum[(size_t)b * Nn + m] = sm;
}

// ---------------------------------------------------------------------------
// Kernel 4: atten_bf16[b,n,m] = exp(scores - colmax[b,m]) / colsum[b,m]
// ---------------------------------------------------------------------------
__global__ void attgru_softmax_kernel(const float* scores, const float* colmax,
                                      const float* colsum, __bf16* atten) {
    size_t idx = (size_t)blockIdx.x * blockDim.x + threadIdx.x;
    size_t total = (size_t)Bn * Nn * Nn;
    if (idx >= total) return;
    int m = (int)(idx % Nn);
    int b = (int)(idx / ((size_t)Nn * Nn));
    float v = __expf(scores[idx] - colmax[(size_t)b * Nn + m]) / colsum[(size_t)b * Nn + m];
    atten[idx] = (__bf16)v;
}

// ---------------------------------------------------------------------------
// Kernel 5: agg[b,t,i] = sum_j atten[b,i,j] * x[b,t,j]  (batched, t-independent)
// grid: (Nn/16, Ss/16, Bn); block: 32
// ---------------------------------------------------------------------------
__global__ __launch_bounds__(32) void attgru_agg_kernel(
        const __bf16* atten, const float* x, float* agg) {
    int lane = threadIdx.x, half = lane >> 4, l16 = lane & 15;
    int i0 = blockIdx.x * 16;
    int t0 = blockIdx.y * 16;
    int b  = blockIdx.z;
    const __bf16* Arow = atten + ((size_t)b * Nn + i0 + l16) * Nn;
    const float*  Bcol = x + ((size_t)b * Ss + t0 + l16) * Nn;   // x row t = B column
    v8f acc = {};
    for (int kb = 0; kb < Nn; kb += 32) {
        v16bf a  = load_a_bf16(Arow + kb, half);
        v16bf bf = load_b_f32cvt(Bcol + kb, half);
        acc = WMMA_BF16(a, bf, acc);
    }
#pragma unroll
    for (int r = 0; r < 8; ++r) {
        int ii = i0 + r + 8 * half;
        int tt = t0 + l16;
        agg[((size_t)b * Ss + tt) * Nn + ii] = acc[r];
    }
}

// ---------------------------------------------------------------------------
// Kernel 6 (x64 steps): one GRU step, K split across 8 waves per block.
//   u_g = h @ Wg^T (M=16 batch, N=2048 nodes, K=2048), g in {r,z,n}
//   wave wv accumulates K-slice [wv*256, wv*256+256); partials -> LDS;
//   then the 256 threads each finish exactly one of the 16x16 tile elements
//   (thread (wv,lane) -> D element (m = wv + 8*(lane>>4), n = lane&15)).
//   grid: Nn/16; block: 256 (8 waves)
// ---------------------------------------------------------------------------
#define STEP_WAVES 8
#define STEP_KSLICE (Nn / STEP_WAVES)   // 256

__global__ __launch_bounds__(32 * STEP_WAVES) void attgru_step_kernel(
        const __bf16* hbf_in, const float* hfp_in,
        const __bf16* Wr, const __bf16* Wz, const __bf16* Wn,
        const float* bhr, const float* bhz, const float* bhn,
        const float* agg, float* hfp_out, __bf16* hbf_out, int t) {
    __shared__ float part[STEP_WAVES][3][8][32];   // 24 KB of the 320 KB WGP LDS

    int tid  = threadIdx.x;
    int wv   = tid >> 5;
    int lane = tid & 31, half = lane >> 4, l16 = lane & 15;
    int i0 = blockIdx.x * 16;
    int k0 = wv * STEP_KSLICE;

    const __bf16* Arow  = hbf_in + (size_t)l16 * Nn + k0;        // batch row slice
    const __bf16* BcolR = Wr + (size_t)(i0 + l16) * Nn + k0;     // weight row = B column
    const __bf16* BcolZ = Wz + (size_t)(i0 + l16) * Nn + k0;
    const __bf16* BcolN = Wn + (size_t)(i0 + l16) * Nn + k0;

    v8f accr = {}, accz = {}, accn = {};
    for (int kb = 0; kb < STEP_KSLICE; kb += 32) {
        if (kb + 32 < STEP_KSLICE) {               // hint next weight chunks
            __builtin_prefetch(BcolR + kb + 32, 0, 1);
            __builtin_prefetch(BcolZ + kb + 32, 0, 1);
            __builtin_prefetch(BcolN + kb + 32, 0, 1);
        }
        v16bf a = load_a_bf16(Arow + kb, half);
        accr = WMMA_BF16(a, load_b_bf16(BcolR + kb, half), accr);
        accz = WMMA_BF16(a, load_b_bf16(BcolZ + kb, half), accz);
        accn = WMMA_BF16(a, load_b_bf16(BcolN + kb, half), accn);
    }
#pragma unroll
    for (int r = 0; r < 8; ++r) {
        part[wv][0][r][lane] = accr[r];
        part[wv][1][r][lane] = accz[r];
        part[wv][2][r][lane] = accn[r];
    }
    __syncthreads();

    // finish one element per thread: r = wv
    float sr = 0.0f, sz = 0.0f, sn = 0.0f;
#pragma unroll
    for (int w = 0; w < STEP_WAVES; ++w) {
        sr += part[w][0][wv][lane];
        sz += part[w][1][wv][lane];
        sn += part[w][2][wv][lane];
    }
    int bb = wv + 8 * half;                  // batch (M dim)
    int ii = i0 + l16;                       // node (N dim)
    float a  = agg[((size_t)bb * Ss + t) * Nn + ii];
    float hp = hfp_in[(size_t)bb * Nn + ii];
    float rg = 1.0f / (1.0f + __expf(-(a + sr + bhr[ii])));
    float zg = 1.0f / (1.0f + __expf(-(a + sz + bhz[ii])));
    float ng = tanhf(a + rg * (sn + bhn[ii]));
    float hn = (1.0f - zg) * ng + zg * hp;
    hfp_out[(size_t)bb * Nn + ii] = hn;
    hbf_out[(size_t)bb * Nn + ii] = (__bf16)hn;
}

// ---------------------------------------------------------------------------
// Kernel 7: out[b] = h[b,:] . Wo[0,:] + bo
// ---------------------------------------------------------------------------
__global__ void attgru_head_kernel(const float* h, const float* Wo, const float* bo, float* out) {
    __shared__ float red[256];
    int b = blockIdx.x, tid = threadIdx.x;
    float s = 0.0f;
    for (int i = tid; i < Nn; i += 256) s += h[(size_t)b * Nn + i] * Wo[i];
    red[tid] = s;
    __syncthreads();
    for (int off = 128; off > 0; off >>= 1) {
        if (tid < off) red[tid] += red[tid + off];
        __syncthreads();
    }
    if (tid == 0) out[b] = red[0] + bo[0];
}

// ---------------------------------------------------------------------------
extern "C" void kernel_launch(void* const* d_in, const int* in_sizes, int n_in,
                              void* d_out, int out_size, void* d_ws, size_t ws_size,
                              hipStream_t stream) {
    (void)in_sizes; (void)n_in; (void)out_size; (void)ws_size;
    const float* x   = (const float*)d_in[0];
    const int*   adj = (const int*)d_in[1];
    const float* Wq  = (const float*)d_in[2];
    const float* bq  = (const float*)d_in[3];
    const float* Wk  = (const float*)d_in[4];
    const float* bk  = (const float*)d_in[5];
    const float* Whr = (const float*)d_in[6];
    const float* bhr = (const float*)d_in[7];
    const float* Whz = (const float*)d_in[8];
    const float* bhz = (const float*)d_in[9];
    const float* Whn = (const float*)d_in[10];
    const float* bhn = (const float*)d_in[11];
    const float* Wo  = (const float*)d_in[12];
    const float* bo  = (const float*)d_in[13];
    float* out = (float*)d_out;

    // ---- carve workspace (256B aligned chunks) ----
    char* w = (char*)d_ws;
    auto carve = [&](size_t bytes) -> char* {
        char* p = w;
        w += (bytes + 255) & ~(size_t)255;
        return p;
    };
    __bf16* Wrb   = (__bf16*)carve((size_t)Nn * Nn * 2);
    __bf16* Wzb   = (__bf16*)carve((size_t)Nn * Nn * 2);
    __bf16* Wnb   = (__bf16*)carve((size_t)Nn * Nn * 2);
    __bf16* Qbf   = (__bf16*)carve((size_t)Bn * Nn * Ee * 2);
    __bf16* Kbf   = (__bf16*)carve((size_t)Bn * Nn * Ee * 2);
    float*  scores= (float*) carve((size_t)Bn * Nn * Nn * 4);
    __bf16* atten = (__bf16*)carve((size_t)Bn * Nn * Nn * 2);
    float*  colmax= (float*) carve((size_t)Bn * Nn * 4);
    float*  colsum= (float*) carve((size_t)Bn * Nn * 4);
    float*  agg   = (float*) carve((size_t)Bn * Ss * Nn * 4);
    float*  hfp[2]; hfp[0] = (float*) carve((size_t)Bn * Nn * 4);
                    hfp[1] = (float*) carve((size_t)Bn * Nn * 4);
    __bf16* hbf[2]; hbf[0] = (__bf16*)carve((size_t)Bn * Nn * 2);
                    hbf[1] = (__bf16*)carve((size_t)Bn * Nn * 2);

    // 0) weight cvt + h0 = 0
    attgru_prep_kernel<<<4096, 256, 0, stream>>>(Whr, Whz, Whn, Wrb, Wzb, Wnb,
                                                 hfp[0], hbf[0]);
    // 1) Q/K projection
    attgru_proj_kernel<<<dim3(Bn * (Nn / 16), Ee / 16, 2), 32, 0, stream>>>(
        x, Wq, bq, Wk, bk, Qbf, Kbf);
    // 2) masked scores
    attgru_scores_kernel<<<dim3(Nn / 16, Nn / 16, Bn), 32, 0, stream>>>(
        Qbf, Kbf, adj, scores);
    // 3) column softmax stats (axis=1)
    attgru_colreduce_kernel<<<Bn * (Nn / 256), 256, 0, stream>>>(scores, colmax, colsum);
    // 4) normalize -> bf16 atten
    {
        size_t total = (size_t)Bn * Nn * Nn;
        attgru_softmax_kernel<<<(unsigned)((total + 255) / 256), 256, 0, stream>>>(
            scores, colmax, colsum, atten);
    }
    // 5) batched aggregate (all timesteps at once)
    attgru_agg_kernel<<<dim3(Nn / 16, Ss / 16, Bn), 32, 0, stream>>>(atten, x, agg);
    // 6) GRU recurrence: 64 sequential steps, ping-pong h buffers
    for (int t = 0; t < Ss; ++t) {
        int p = t & 1;
        attgru_step_kernel<<<Nn / 16, 32 * STEP_WAVES, 0, stream>>>(
            hbf[p], hfp[p], Wrb, Wzb, Wnb, bhr, bhz, bhn, agg,
            hfp[1 - p], hbf[1 - p], t);
    }
    // final h lands in buffer 0 (t=63 reads 1, writes 0)
    // 7) output head
    attgru_head_kernel<<<Bn, 256, 0, stream>>>(hfp[0], Wo, bo, out);
}
